// ModifiedShallow_60284160966841
// MI455X (gfx1250) — compile-verified, matrix operand order checked by reading, not run on hardware
//
#include <hip/hip_runtime.h>

// ---------- types ----------
typedef __attribute__((ext_vector_type(16))) __bf16 v16bf;
typedef __attribute__((ext_vector_type(8)))  float  v8f;
typedef __attribute__((ext_vector_type(4)))  int    v4i;

struct Frag32B { uint4 lo; uint4 hi; };   // 32 bytes == v16bf

static __device__ __forceinline__ unsigned short f32_to_bf16(float f) {
    unsigned int u = __float_as_uint(f);
    u += 0x7FFFu + ((u >> 16) & 1u);       // round-to-nearest-even
    return (unsigned short)(u >> 16);
}

// Async global->LDS staging (gfx1250 GLOBAL_LOAD_ASYNC_TO_LDS_B128, ASYNCcnt).
// Signature (from clang diagnostic): (int4 AS1*, int4 AS3*, imm offset, imm cpol)
#if defined(__gfx1250__) && __has_builtin(__builtin_amdgcn_global_load_async_to_lds_b128)
#define GEMM_ASYNC 1
#else
#define GEMM_ASYNC 0
#endif

#if GEMM_ASYNC
typedef __attribute__((address_space(1))) v4i gv4i;
typedef __attribute__((address_space(3))) v4i lv4i;

static __device__ __forceinline__ void cp16_async(const unsigned short* gsrc,
                                                  unsigned short* ldst) {
    __builtin_amdgcn_global_load_async_to_lds_b128(
        (gv4i*)(v4i*)const_cast<unsigned short*>(gsrc),
        (lv4i*)(v4i*)ldst, 0, 0);
}
static __device__ __forceinline__ void wait_async0() {
#if __has_builtin(__builtin_amdgcn_s_wait_asynccnt)
    __builtin_amdgcn_s_wait_asynccnt(0);
#else
    asm volatile("s_wait_asynccnt 0x0" ::: "memory");
#endif
}
#endif

// ---------- fp32 -> bf16 conversion (vectorized x4) ----------
__global__ __launch_bounds__(256)
void cvt_f32_bf16(const float* __restrict__ in, unsigned short* __restrict__ out, int n4) {
    int i = blockIdx.x * blockDim.x + threadIdx.x;
    if (i >= n4) return;
    float4 f = ((const float4*)in)[i];
    ushort4 o;
    o.x = f32_to_bf16(f.x);
    o.y = f32_to_bf16(f.y);
    o.z = f32_to_bf16(f.z);
    o.w = f32_to_bf16(f.w);
    ((ushort4*)out)[i] = o;
}

// ---------- fp32 -> bf16 with transpose: in[K][N] f32 -> out[N][K] bf16 ----------
// 32x32 tiles via LDS; one-time cost so B-operands of every GEMM are K-contiguous.
__global__ __launch_bounds__(256)
void cvt_transpose_f32_bf16(const float* __restrict__ in, unsigned short* __restrict__ out,
                            int K, int N) {
    __shared__ unsigned short tile[32][33];
    const int n0 = blockIdx.x * 32;
    const int k0 = blockIdx.y * 32;
    const int tx = threadIdx.x & 31;
    const int ty = threadIdx.x >> 5;       // 0..7
#pragma unroll
    for (int i = 0; i < 4; ++i) {
        int k = ty + i * 8;
        tile[k][tx] = f32_to_bf16(in[(size_t)(k0 + k) * N + n0 + tx]);
    }
    __syncthreads();
#pragma unroll
    for (int i = 0; i < 4; ++i) {
        int n = ty + i * 8;
        out[(size_t)(n0 + n) * K + k0 + tx] = tile[tx][n];
    }
}

// ---------- bf16 WMMA GEMM: C[M,N] = act(A[M,K] @ Bt[N,K]^T + bias) ----------
// block tile 128x128, 256 threads = 8 waves (4 in M x 2 in N),
// wave tile 32x64 = 2x4 wmma 16x16 tiles, K-step 32, double-buffered LDS.
template<bool RELU, bool OUT_F32>
__global__ __launch_bounds__(256)
void gemm_bf16_wmma(const unsigned short* __restrict__ A,    // [M][K] bf16
                    const unsigned short* __restrict__ Bt,   // [N][K] bf16 (pre-transposed)
                    const float* __restrict__ bias,
                    void* __restrict__ Cout,
                    int M, int N, int K) {
    __shared__ __align__(16) unsigned short sA[2][128 * 32]; // [buf][m][k]  2 x 8 KB
    __shared__ __align__(16) unsigned short sB[2][128 * 32]; // [buf][n][k]  2 x 8 KB

    const int tid  = threadIdx.x;
    const int wave = tid >> 5;
    const int lane = tid & 31;
    const int half = lane >> 4;
    const int l16  = lane & 15;
    const int wm   = wave & 3;      // wave position in M (0..3)
    const int wn   = wave >> 2;     // wave position in N (0..1)
    const int bm   = blockIdx.y * 128;
    const int bn   = blockIdx.x * 128;

    // staging: per tile 512 chunks of 16B; each thread owns rows r0 and r0+64
    const int r0 = tid >> 2;            // 0..63
    const int cc = (tid & 3) * 8;       // element offset of this thread's 16B chunk

    const unsigned short* gA0 = A  + (size_t)(bm + r0) * K + cc;
    const unsigned short* gA1 = gA0 + (size_t)64 * K;
    const unsigned short* gB0 = Bt + (size_t)(bn + r0) * K + cc;
    const unsigned short* gB1 = gB0 + (size_t)64 * K;
    const int lo0 = r0 * 32 + cc;
    const int lo1 = (r0 + 64) * 32 + cc;

    // ---- prologue: stage tile 0 into buffer 0 ----
#if GEMM_ASYNC
    cp16_async(gA0, &sA[0][lo0]);
    cp16_async(gA1, &sA[0][lo1]);
    cp16_async(gB0, &sB[0][lo0]);
    cp16_async(gB1, &sB[0][lo1]);
    wait_async0();
#else
    *(uint4*)&sA[0][lo0] = *(const uint4*)gA0;
    *(uint4*)&sA[0][lo1] = *(const uint4*)gA1;
    *(uint4*)&sB[0][lo0] = *(const uint4*)gB0;
    *(uint4*)&sB[0][lo1] = *(const uint4*)gB1;
#endif
    __syncthreads();

    v8f acc[2][4];
#pragma unroll
    for (int mi = 0; mi < 2; ++mi)
#pragma unroll
        for (int ni = 0; ni < 4; ++ni)
            acc[mi][ni] = (v8f){0.f, 0.f, 0.f, 0.f, 0.f, 0.f, 0.f, 0.f};

    const int nk = K >> 5;
    for (int kt = 0; kt < nk; ++kt) {
        const int cur  = kt & 1;
        const int nxt  = cur ^ 1;
        const int koff = (kt + 1) << 5;       // element offset of next K-tile
        const bool more = (kt + 1 < nk);

        // ---- issue staging for next tile (overlaps with WMMAs below) ----
#if GEMM_ASYNC
        if (more) {
            cp16_async(gA0 + koff, &sA[nxt][lo0]);
            cp16_async(gA1 + koff, &sA[nxt][lo1]);
            cp16_async(gB0 + koff, &sB[nxt][lo0]);
            cp16_async(gB1 + koff, &sB[nxt][lo1]);
        }
#else
        uint4 a0 = {}, a1 = {}, b0 = {}, b1 = {};
        if (more) {
            a0 = *(const uint4*)(gA0 + koff);
            a1 = *(const uint4*)(gA1 + koff);
            b0 = *(const uint4*)(gB0 + koff);
            b1 = *(const uint4*)(gB1 + koff);
        }
#endif

        // ---- compute on current buffer ----
        const unsigned short* cA = sA[cur];
        const unsigned short* cB = sB[cur];

        v16bf afrag[2];
#pragma unroll
        for (int mi = 0; mi < 2; ++mi) {
            int r = wm * 32 + mi * 16 + l16;
            Frag32B f;
            f.lo = *(const uint4*)(cA + r * 32 + half * 8);
            f.hi = *(const uint4*)(cA + r * 32 + 16 + half * 8);
            afrag[mi] = __builtin_bit_cast(v16bf, f);
        }
#pragma unroll
        for (int ni = 0; ni < 4; ++ni) {
            int n = wn * 64 + ni * 16 + l16;
            Frag32B f;
            f.lo = *(const uint4*)(cB + n * 32 + half * 8);
            f.hi = *(const uint4*)(cB + n * 32 + 16 + half * 8);
            v16bf bfrag = __builtin_bit_cast(v16bf, f);
#pragma unroll
            for (int mi = 0; mi < 2; ++mi)
                acc[mi][ni] = __builtin_amdgcn_wmma_f32_16x16x32_bf16(
                    false, afrag[mi], false, bfrag,
                    (short)0, acc[mi][ni], false, false);
        }

        // ---- commit next tile, one barrier per iteration ----
#if GEMM_ASYNC
        wait_async0();
#else
        if (more) {
            *(uint4*)&sA[nxt][lo0] = a0;
            *(uint4*)&sA[nxt][lo1] = a1;
            *(uint4*)&sB[nxt][lo0] = b0;
            *(uint4*)&sB[nxt][lo1] = b1;
        }
#endif
        __syncthreads();
    }

    // ---- epilogue: bias (+ReLU), store bf16 or f32 ----
#pragma unroll
    for (int mi = 0; mi < 2; ++mi) {
#pragma unroll
        for (int ni = 0; ni < 4; ++ni) {
            int col = bn + wn * 64 + ni * 16 + l16;
            float bv = bias[col];
#pragma unroll
            for (int r = 0; r < 8; ++r) {
                int row = bm + wm * 32 + mi * 16 + half * 8 + r;
                float v = acc[mi][ni][r] + bv;
                if (RELU) v = v > 0.f ? v : 0.f;
                if (OUT_F32)
                    ((float*)Cout)[(size_t)row * N + col] = v;
                else
                    ((unsigned short*)Cout)[(size_t)row * N + col] = f32_to_bf16(v);
            }
        }
    }
}

// ---------- out[n] = sum_k logits[n,k] * w[k,n]  (one wave per row) ----------
__global__ __launch_bounds__(256)
void rowdot_kernel(const float* __restrict__ logits, const float* __restrict__ w,
                   float* __restrict__ out, int N, int K) {
    int row  = (blockIdx.x * blockDim.x + threadIdx.x) >> 5;
    int lane = threadIdx.x & 31;
    if (row >= N) return;
    float s = 0.f;
    for (int k = lane; k < K; k += 32)
        s += logits[(size_t)row * K + k] * w[(size_t)k * N + row];
#pragma unroll
    for (int off = 16; off > 0; off >>= 1)
        s += __shfl_down(s, off, 32);
    if (lane == 0) out[row] = s;
}

// ---------- launcher ----------
extern "C" void kernel_launch(void* const* d_in, const int* in_sizes, int n_in,
                              void* d_out, int out_size, void* d_ws, size_t ws_size,
                              hipStream_t stream) {
    (void)in_sizes; (void)n_in; (void)out_size; (void)ws_size;

    const int N_  = 4096;
    const int Din = 1024, H1 = 2048, H2 = 2048, H3 = 1024, Dout = 512;

    const float* features = (const float*)d_in[0];
    const float* W1 = (const float*)d_in[1];
    const float* b1 = (const float*)d_in[2];
    const float* W2 = (const float*)d_in[3];
    const float* b2 = (const float*)d_in[4];
    const float* W3 = (const float*)d_in[5];
    const float* b3 = (const float*)d_in[6];
    const float* Wh = (const float*)d_in[7];
    const float* bh = (const float*)d_in[8];
    const float* w  = (const float*)d_in[9];

    // workspace layout (lifetime-aliased)
    char* ws = (char*)d_ws;
    size_t off = 0;
    auto alloc = [&](size_t bytes) {
        void* p = ws + off;
        off += (bytes + 255) & ~(size_t)255;
        return p;
    };
    unsigned short* Xbf  = (unsigned short*)alloc((size_t)N_  * Din * 2); // aliased: h3 later
    unsigned short* h1   = (unsigned short*)alloc((size_t)N_  * H1  * 2);
    unsigned short* h2   = (unsigned short*)alloc((size_t)N_  * H2  * 2);
    unsigned short* W1t  = (unsigned short*)alloc((size_t)Din * H1  * 2); // [H1][Din]
    unsigned short* W2t  = (unsigned short*)alloc((size_t)H1  * H2  * 2); // [H2][H1], aliased: logits
    unsigned short* W3t  = (unsigned short*)alloc((size_t)H2  * H3  * 2); // [H3][H2]
    unsigned short* Wht  = (unsigned short*)alloc((size_t)H3  * Dout * 2);// [Dout][H3]
    unsigned short* h3   = Xbf;               // Xbf dead after layer 1
    float*          logits = (float*)W2t;     // W2t dead after layer 2 (8MB each)

    // features: plain f32->bf16
    {
        int n4 = (N_ * Din) / 4;
        hipLaunchKernelGGL(cvt_f32_bf16, dim3((n4 + 255) / 256), dim3(256), 0, stream,
                           features, Xbf, n4);
    }
    // weights: f32->bf16 with transpose (out[N][K] from in[K][N])
    auto cvtT = [&](const float* src, unsigned short* dst, int K, int N) {
        hipLaunchKernelGGL(cvt_transpose_f32_bf16, dim3(N / 32, K / 32), dim3(256), 0, stream,
                           src, dst, K, N);
    };
    cvtT(W1, W1t, Din, H1);
    cvtT(W2, W2t, H1,  H2);
    cvtT(W3, W3t, H2,  H3);
    cvtT(Wh, Wht, H3,  Dout);

    // L1: h1 = relu(X @ W1 + b1)   [4096,2048]
    hipLaunchKernelGGL((gemm_bf16_wmma<true, false>),
                       dim3(H1 / 128, N_ / 128), dim3(256), 0, stream,
                       Xbf, W1t, b1, (void*)h1, N_, H1, Din);
    // L2: h2 = relu(h1 @ W2 + b2)  [4096,2048]
    hipLaunchKernelGGL((gemm_bf16_wmma<true, false>),
                       dim3(H2 / 128, N_ / 128), dim3(256), 0, stream,
                       h1, W2t, b2, (void*)h2, N_, H2, H1);
    // L3: h3 = h2 @ W3 + b3        [4096,1024] (no relu)
    hipLaunchKernelGGL((gemm_bf16_wmma<false, false>),
                       dim3(H3 / 128, N_ / 128), dim3(256), 0, stream,
                       h2, W3t, b3, (void*)h3, N_, H3, H2);
    // L4: logits = h3 @ Wh + bh    [4096,512] f32 out
    hipLaunchKernelGGL((gemm_bf16_wmma<false, true>),
                       dim3(Dout / 128, N_ / 128), dim3(256), 0, stream,
                       h3, Wht, bh, (void*)logits, N_, Dout, H3);
    // out[n] = sum_k logits[n,k] * w[k,n]
    hipLaunchKernelGGL(rowdot_kernel, dim3(N_ / 8), dim3(256), 0, stream,
                       logits, w, (float*)d_out, N_, Dout);
}
